// HeterogeneousGINLayer_81552839016473
// MI455X (gfx1250) — compile-verified, matrix operand order checked by reading, not run on hardware
//
#include <hip/hip_runtime.h>
#include <hip/hip_bf16.h>

// ---------------------------------------------------------------------------
// HeteroGIN on MI455X (gfx1250, wave32).
// GEMMs run on v_wmma_f32_16x16x32_bf16 (bf16 A/B, fp32 accumulate) with a
// double-buffered LDS pipeline: global->reg loads for K-step k+1 are issued
// before the WMMAs of step k, fragments are fed from LDS (ds_load_b128), and
// __syncthreads (s_wait_dscnt + barrier) rotates the buffers. This hides L2
// latency structurally instead of relying on the register allocator.
// ---------------------------------------------------------------------------

#define NN   50000   // nodes per type
#define DD   512     // channels
#define EE   150000  // edges per type
#define LSTR 40      // LDS row stride in halves (80 B = 20 banks, pad vs 64B)

typedef __attribute__((ext_vector_type(16))) __bf16 v16bf;
typedef __attribute__((ext_vector_type(8)))  __bf16 v8bf;
typedef __attribute__((ext_vector_type(8)))  float  v8f;
typedef __attribute__((ext_vector_type(4)))  float  v4f;

// ---------------------------------------------------------------- zero agg
__global__ void hg_zero_f32(float* __restrict__ p, long long n4) {
  long long i = (long long)blockIdx.x * blockDim.x + threadIdx.x;
  if (i < n4) {
    v4f z = {0.f, 0.f, 0.f, 0.f};
    ((v4f*)p)[i] = z;
  }
}

// ------------------------------------------------------- edge scatter-add
__global__ void hg_scatter_add(const float* __restrict__ xsrc,
                               const int* __restrict__ ei,
                               float* __restrict__ agg) {
  long long idx = (long long)blockIdx.x * blockDim.x + threadIdx.x;
  int e = (int)(idx >> 7);          // DD/4 = 128 chunks per edge
  int c = ((int)idx & 127) << 2;
  if (e >= EE) return;
  int s = ei[e];
  int d = ei[EE + e];
  v4f v = *(const v4f*)(xsrc + (long long)s * DD + c);
  float* out = agg + (long long)d * DD + c;
  __hip_atomic_fetch_add(out + 0, v.x, __ATOMIC_RELAXED, __HIP_MEMORY_SCOPE_AGENT);
  __hip_atomic_fetch_add(out + 1, v.y, __ATOMIC_RELAXED, __HIP_MEMORY_SCOPE_AGENT);
  __hip_atomic_fetch_add(out + 2, v.z, __ATOMIC_RELAXED, __HIP_MEMORY_SCOPE_AGENT);
  __hip_atomic_fetch_add(out + 3, v.w, __ATOMIC_RELAXED, __HIP_MEMORY_SCOPE_AGENT);
}

// ------------------------------------------- weight fp32[K][N] -> bf16[N][K]
__global__ void hg_wt_convert(const float* __restrict__ W,
                              __bf16* __restrict__ Wt) {
  __shared__ __bf16 tile[32][33];
  const int tx = threadIdx.x;      // 32
  const int ty = threadIdx.y;      // 8
  const int n0 = blockIdx.x * 32;
  const int k0 = blockIdx.y * 32;
#pragma unroll
  for (int i = 0; i < 4; ++i) {
    int k = k0 + ty + i * 8;
    tile[ty + i * 8][tx] = (__bf16)W[(long long)k * DD + n0 + tx];
  }
  __syncthreads();
#pragma unroll
  for (int i = 0; i < 4; ++i) {
    int n = n0 + ty + i * 8;
    Wt[(long long)n * DD + k0 + tx] = tile[tx][ty + i * 8];
  }
}

// ------------------------------------------------- A = bf16(x_dst + agg)
__global__ void hg_prep_a(const float* __restrict__ x,
                          const float* __restrict__ agg,
                          __bf16* __restrict__ A) {
  long long i = (long long)blockIdx.x * blockDim.x + threadIdx.x;
  if (i >= (long long)NN * (DD / 4)) return;
  long long base = i << 2;
  v4f xv = *(const v4f*)(x + base);
  v4f av = *(const v4f*)(agg + base);
  union { __bf16 h[4]; uint2 u; } r;
  r.h[0] = (__bf16)(xv.x + av.x);
  r.h[1] = (__bf16)(xv.y + av.y);
  r.h[2] = (__bf16)(xv.z + av.z);
  r.h[3] = (__bf16)(xv.w + av.w);
  *(uint2*)(A + base) = r.u;
}

// --------------------------------------------------------------- WMMA GEMM
// C[M,512] = A[M,512] @ W[512,512] (+bias).  A bf16 row-major, Wt bf16 [N][K].
// Block: 256 thr = 8 waves (2 M x 4 N); block tile 32x256; K-step 32.
// Double-buffered LDS staging; fragments read via ds_load_b128.
// mode 0: relu -> bf16 (hidden);  mode 1: fp32 store;  mode 2: fp32 +=.
__global__ __launch_bounds__(256)
void hg_gemm_bf16(const __bf16* __restrict__ A,
                  const __bf16* __restrict__ Wt,
                  const float* __restrict__ bias,
                  float* __restrict__ outf,
                  __bf16* __restrict__ outh,
                  int Mtot, int mode) {
  __shared__ __align__(16) __bf16 ldsA[2][32 * LSTR];    //  2 x 2.5 KB
  __shared__ __align__(16) __bf16 ldsB[2][256 * LSTR];   //  2 x 20  KB

  const int tid   = threadIdx.x;
  const int lane  = tid & 31;
  const int wave  = tid >> 5;
  const int waveM = wave >> 2;               // 0..1
  const int waveN = wave & 3;                // 0..3
  const int m0blk = blockIdx.x * 32;
  const int n0blk = blockIdx.y * 256;
  const int lm    = lane & 15;
  const int khalf = lane >> 4;               // ISA 16-bit A/B lane split

  // ---- staging roles ------------------------------------------------------
  // B: thread tid owns Wt row (n0blk + tid), copies 32 halves (4 x 16B).
  const __bf16* bgp = Wt + (long long)(n0blk + tid) * DD;
  // A: threads 0..127, (row, seg) = (tid>>2, tid&3), copy 8 halves each.
  const int arowi = tid >> 2;
  const int aseg  = tid & 3;
  int agrow = m0blk + arowi;
  if (agrow >= Mtot) agrow = Mtot - 1;       // pad rows read a valid row
  const __bf16* agp = A + (long long)agrow * DD;

  v8bf sB[4];
  v8bf sA;

  // ---- prologue: stage K-step 0 ------------------------------------------
#pragma unroll
  for (int c = 0; c < 4; ++c) sB[c] = *(const v8bf*)(bgp + c * 8);
  if (tid < 128) sA = *(const v8bf*)(agp + aseg * 8);
#pragma unroll
  for (int c = 0; c < 4; ++c)
    *(v8bf*)&ldsB[0][tid * LSTR + c * 8] = sB[c];
  if (tid < 128) *(v8bf*)&ldsA[0][arowi * LSTR + aseg * 8] = sA;
  __syncthreads();

  // ---- compute-side LDS offsets (halves) ---------------------------------
  const int aoff     = (waveM * 16 + lm) * LSTR + khalf * 8;
  const int boffbase = (waveN * 64 + lm) * LSTR + khalf * 8;

  v8f acc[4];
#pragma unroll
  for (int t = 0; t < 4; ++t) {
    v8f z = {0.f, 0.f, 0.f, 0.f, 0.f, 0.f, 0.f, 0.f};
    acc[t] = z;
  }

#pragma unroll
  for (int k0 = 0; k0 < DD; k0 += 32) {
    const int kn  = k0 + 32;
    const int cur = (k0 >> 5) & 1;
    const int nxt = cur ^ 1;

    // issue next K-step's global loads first; they fly during the WMMAs
    if (kn < DD) {
#pragma unroll
      for (int c = 0; c < 4; ++c) sB[c] = *(const v8bf*)(bgp + kn + c * 8);
      if (tid < 128) sA = *(const v8bf*)(agp + kn + aseg * 8);
      __builtin_prefetch(bgp + kn + 32, 0, 1);   // global_prefetch_b8
    }

    // fragments from LDS (short latency), then 4 WMMAs
    v8bf alo = *(const v8bf*)&ldsA[cur][aoff];
    v8bf ahi = *(const v8bf*)&ldsA[cur][aoff + 16];
    v16bf afrag = __builtin_shufflevector(alo, ahi,
        0, 1, 2, 3, 4, 5, 6, 7, 8, 9, 10, 11, 12, 13, 14, 15);
#pragma unroll
    for (int t = 0; t < 4; ++t) {
      const int boff = boffbase + t * 16 * LSTR;
      v8bf blo = *(const v8bf*)&ldsB[cur][boff];
      v8bf bhi = *(const v8bf*)&ldsB[cur][boff + 16];
      v16bf bfrag = __builtin_shufflevector(blo, bhi,
          0, 1, 2, 3, 4, 5, 6, 7, 8, 9, 10, 11, 12, 13, 14, 15);
      acc[t] = __builtin_amdgcn_wmma_f32_16x16x32_bf16(
          false, afrag, false, bfrag, (short)0, acc[t], false, false);
    }

    // rotate buffers: store staged regs, make visible to all waves
    if (kn < DD) {
#pragma unroll
      for (int c = 0; c < 4; ++c)
        *(v8bf*)&ldsB[nxt][tid * LSTR + c * 8] = sB[c];
      if (tid < 128) *(v8bf*)&ldsA[nxt][arowi * LSTR + aseg * 8] = sA;
      __syncthreads();
    }
  }

  // Epilogue: C/D layout — lane gives N, reg r gives M = m0 + khalf*8 + r.
  const int rbase = m0blk + waveM * 16 + khalf * 8;
#pragma unroll
  for (int t = 0; t < 4; ++t) {
    int ncol = n0blk + waveN * 64 + t * 16 + lm;
    float b = bias[ncol];
#pragma unroll
    for (int r = 0; r < 8; ++r) {
      int row = rbase + r;
      if (row < Mtot) {
        float v = acc[t][r] + b;
        long long o = (long long)row * DD + ncol;
        if (mode == 0) {
          v = v > 0.f ? v : 0.f;
          outh[o] = (__bf16)v;
        } else if (mode == 1) {
          outf[o] = v;
        } else {
          outf[o] += v;
        }
      }
    }
  }
}

// ---------------------------------------------------------------------------
extern "C" void kernel_launch(void* const* d_in, const int* in_sizes, int n_in,
                              void* d_out, int out_size, void* d_ws, size_t ws_size,
                              hipStream_t stream) {
  (void)in_sizes; (void)n_in; (void)out_size; (void)ws_size;

  const float* x_a = (const float*)d_in[0];
  const float* x_b = (const float*)d_in[1];
  const int* ei_aa = (const int*)d_in[2];
  const int* ei_ab = (const int*)d_in[3];
  const int* ei_ba = (const int*)d_in[4];
  const float* w1_aa = (const float*)d_in[5];
  const float* b1_aa = (const float*)d_in[6];
  const float* w2_aa = (const float*)d_in[7];
  const float* b2_aa = (const float*)d_in[8];
  const float* w1_ab = (const float*)d_in[9];
  const float* b1_ab = (const float*)d_in[10];
  const float* w2_ab = (const float*)d_in[11];
  const float* b2_ab = (const float*)d_in[12];
  const float* w1_ba = (const float*)d_in[13];
  const float* b1_ba = (const float*)d_in[14];
  const float* w2_ba = (const float*)d_in[15];
  const float* b2_ba = (const float*)d_in[16];

  float* out_a = (float*)d_out;
  float* out_b = out_a + (size_t)NN * DD;

  // workspace carving (all offsets 256B-aligned)
  char* ws = (char*)d_ws;
  const size_t SZF = (size_t)NN * DD * sizeof(float);   // 102.4 MB
  const size_t SZH = (size_t)NN * DD * sizeof(__bf16);  //  51.2 MB
  const size_t SZW = (size_t)DD * DD * sizeof(__bf16);  //   0.5 MB
  float*  agg_aa = (float*)(ws);
  float*  agg_ab = (float*)(ws + SZF);
  float*  agg_ba = (float*)(ws + 2 * SZF);
  __bf16* Abf    = (__bf16*)(ws + 3 * SZF);
  __bf16* hidden = (__bf16*)(ws + 3 * SZF + SZH);
  __bf16* Wt[6];
  for (int i = 0; i < 6; ++i) Wt[i] = (__bf16*)(ws + 3 * SZF + 2 * SZH + i * SZW);

  // 1. zero agg slabs
  {
    long long n4 = (3LL * NN * DD) / 4;
    hg_zero_f32<<<(unsigned)((n4 + 255) / 256), 256, 0, stream>>>(agg_aa, n4);
  }

  // 2. edge scatter per edge type (src per reference: aa: a, ab: a, ba: b)
  {
    unsigned blk = (unsigned)(((long long)EE * 128 + 255) / 256);
    hg_scatter_add<<<blk, 256, 0, stream>>>(x_a, ei_aa, agg_aa);
    hg_scatter_add<<<blk, 256, 0, stream>>>(x_a, ei_ab, agg_ab);
    hg_scatter_add<<<blk, 256, 0, stream>>>(x_b, ei_ba, agg_ba);
  }

  // 3. weights -> bf16 [N][K]
  {
    dim3 g(DD / 32, DD / 32), b(32, 8);
    hg_wt_convert<<<g, b, 0, stream>>>(w1_aa, Wt[0]);
    hg_wt_convert<<<g, b, 0, stream>>>(w2_aa, Wt[1]);
    hg_wt_convert<<<g, b, 0, stream>>>(w1_ab, Wt[2]);
    hg_wt_convert<<<g, b, 0, stream>>>(w2_ab, Wt[3]);
    hg_wt_convert<<<g, b, 0, stream>>>(w1_ba, Wt[4]);
    hg_wt_convert<<<g, b, 0, stream>>>(w2_ba, Wt[5]);
  }

  const unsigned prepBlk = (unsigned)(((long long)NN * (DD / 4) + 255) / 256);
  const dim3 ggrid((NN + 31) / 32, DD / 256);  // 1563 x 2

  // 4. conv aa: h_a  = mlp_aa(x_a + agg_aa)
  hg_prep_a<<<prepBlk, 256, 0, stream>>>(x_a, agg_aa, Abf);
  hg_gemm_bf16<<<ggrid, 256, 0, stream>>>(Abf, Wt[0], b1_aa, nullptr, hidden, NN, 0);
  hg_gemm_bf16<<<ggrid, 256, 0, stream>>>(hidden, Wt[1], b2_aa, out_a, nullptr, NN, 1);

  // 5. conv ba: h_a += mlp_ba(x_a + agg_ba)
  hg_prep_a<<<prepBlk, 256, 0, stream>>>(x_a, agg_ba, Abf);
  hg_gemm_bf16<<<ggrid, 256, 0, stream>>>(Abf, Wt[4], b1_ba, nullptr, hidden, NN, 0);
  hg_gemm_bf16<<<ggrid, 256, 0, stream>>>(hidden, Wt[5], b2_ba, out_a, nullptr, NN, 2);

  // 6. conv ab: h_b  = mlp_ab(x_b + agg_ab)
  hg_prep_a<<<prepBlk, 256, 0, stream>>>(x_b, agg_ab, Abf);
  hg_gemm_bf16<<<ggrid, 256, 0, stream>>>(Abf, Wt[2], b1_ab, nullptr, hidden, NN, 0);
  hg_gemm_bf16<<<ggrid, 256, 0, stream>>>(hidden, Wt[3], b2_ab, out_b, nullptr, NN, 1);
}
